// NeuralMemory_5523327943065
// MI455X (gfx1250) — compile-verified, matrix operand order checked by reading, not run on hardware
//
#include <hip/hip_runtime.h>
#include <math.h>

// ---------------- problem constants ----------------
#define DIMM   512
#define HEADS  4
#define DHH    128
#define HIDD   512
#define CHK    64
#define BBATCH 2
#define NTOK   2048
#define NCHNK  32           // chunks per (b,h)
#define NMAT   (DHH*HIDD)   // 65536 elems per fast-weight matrix

typedef __bf16 bf16x16 __attribute__((ext_vector_type(16)));
typedef float  f32x8   __attribute__((ext_vector_type(8)));

// ---------------- workspace layout (float offsets) ----------------
static const size_t OFF_SSTORE = 0;                       // 2*2048*512
static const size_t OFF_SRETR  = 2097152;
static const size_t OFF_K      = 4194304;
static const size_t OFF_V      = 6291456;
static const size_t OFF_Q      = 8388608;
static const size_t OFF_LR     = 10485760;                // (4096,4)
static const size_t OFF_GATE   = 10502144;                // (4096,4)
static const size_t OFF_POOLED = 10518528;                // (64,512)
static const size_t OFF_MOM    = 10551296;                // (b,h,nc)=256
static const size_t OFF_DEC    = 10551552;                // 256
static const size_t OFF_MW2T   = 10551808;                // (4,128,512)
static const size_t OFF_GG     = 10813952;                // (256,128)
static const size_t OFF_H1     = 10846720;                // 256*64*512  (also NS ping buffer T2 start)
static const size_t OFF_A1     = 19235328;                // 256*64*512  (T2 spans H1..A1)
static const size_t OFF_H2     = 27623936;                // 256*64*128
static const size_t OFF_DH2    = 29721088;                // 256*64*128
static const size_t OFF_ABUF   = 31818240;                // 256*128*128
static const size_t OFF_BM     = 36012544;                // 256*128*128
static const size_t OFF_S12    = 40206848;                // 512*65536 (g1 half then g2t half)
static const size_t OFF_FINAL  = 73761280;                // (4096,512)
static const size_t WS_FLOATS  = 75858432;

// ---------------- generic batched WMMA GEMM (bf16 inputs, f32 accum) ----------------
// C[b,h,c] = alpha * op(A) @ op(B) + beta * Cin ; batch z -> (b,h,c).
// Each wave computes a 32x32 C tile: 2 A frags x 2 B frags -> 4 WMMAs per 32-K step,
// so each loaded fragment is reused twice (2x arithmetic intensity vs 16x32).
// All per-lane offsets are 32-bit (buffers < 2^31 elements).
template<int TA, int TB>
__global__ __launch_bounds__(256) void gemm_bf16(
    int M, int N, int Kd,
    const float* __restrict__ A, int lda, int sAb, int sAh, int sAc,
    const float* __restrict__ B, int ldb, int sBb, int sBh, int sBc,
    float* __restrict__ C, int ldc, int sCb, int sCh, int sCc,
    const float* __restrict__ Cin, float alpha, float beta, int nh, int nc)
{
    int z = blockIdx.z;
    int c = z % nc;
    int h = (z / nc) % nh;
    int b = z / (nc * nh);
    const float* Ab = A + (long)b*sAb + (long)h*sAh + (long)c*sAc;
    const float* Bb = B + (long)b*sBb + (long)h*sBh + (long)c*sBc;
    float*       Cb = C + (long)b*sCb + (long)h*sCh + (long)c*sCc;
    const float* Ci = Cin ? (Cin + (long)b*sCb + (long)h*sCh + (long)c*sCc) : nullptr;

    int lane = threadIdx.x & 31;
    int wave = threadIdx.x >> 5;
    int mr  = lane & 15;
    int hlf = lane >> 4;
    int tiles_n2 = N >> 5;                 // 32-wide tile columns
    int tile = blockIdx.y * 8 + wave;
    int m0 = (tile / tiles_n2) << 5;       // 32-high tile rows
    int n0 = (tile % tiles_n2) << 5;

    // K offsets within a 32-step for the 8 dwords of a 16-bit fragment (ISA 7.12.2)
    int kbo[8];
    #pragma unroll
    for (int r = 0; r < 8; ++r)
        kbo[r] = (r < 4) ? (hlf*8 + 2*r) : (16 + hlf*8 + 2*(r-4));

    const float* ap0 = TA ? (Ab + m0 + mr)         : (Ab + (m0+mr)*lda);
    const float* ap1 = TA ? (Ab + m0 + 16 + mr)    : (Ab + (m0+16+mr)*lda);
    const float* bp0 = TB ? (Bb + (n0+mr)*ldb)     : (Bb + n0 + mr);
    const float* bp1 = TB ? (Bb + (n0+16+mr)*ldb)  : (Bb + n0 + 16 + mr);

    f32x8 acc00 = {0.f,0.f,0.f,0.f,0.f,0.f,0.f,0.f};
    f32x8 acc01 = acc00, acc10 = acc00, acc11 = acc00;

    for (int k0 = 0; k0 < Kd; k0 += 32) {
        // prefetch the streaming (contiguous) operands two K-steps ahead
        int kp = (k0 + 64 < Kd) ? (k0 + 64) : k0;   // uniform select
        if (!TA) { __builtin_prefetch(ap0 + kp, 0, 0); __builtin_prefetch(ap1 + kp, 0, 0); }
        if (TB)  { __builtin_prefetch(bp0 + kp, 0, 0); __builtin_prefetch(bp1 + kp, 0, 0); }

        bf16x16 af0, af1, bf0, bf1;
        #pragma unroll
        for (int r = 0; r < 8; ++r) {
            int kk = k0 + kbo[r];
            if (TA) {
                af0[2*r]   = (__bf16)ap0[kk*lda];     af0[2*r+1] = (__bf16)ap0[(kk+1)*lda];
                af1[2*r]   = (__bf16)ap1[kk*lda];     af1[2*r+1] = (__bf16)ap1[(kk+1)*lda];
            } else {
                float2 v0 = *(const float2*)(ap0 + kk);
                float2 v1 = *(const float2*)(ap1 + kk);
                af0[2*r] = (__bf16)v0.x; af0[2*r+1] = (__bf16)v0.y;
                af1[2*r] = (__bf16)v1.x; af1[2*r+1] = (__bf16)v1.y;
            }
            if (TB) {
                float2 u0 = *(const float2*)(bp0 + kk);
                float2 u1 = *(const float2*)(bp1 + kk);
                bf0[2*r] = (__bf16)u0.x; bf0[2*r+1] = (__bf16)u0.y;
                bf1[2*r] = (__bf16)u1.x; bf1[2*r+1] = (__bf16)u1.y;
            } else {
                bf0[2*r]   = (__bf16)bp0[kk*ldb];     bf0[2*r+1] = (__bf16)bp0[(kk+1)*ldb];
                bf1[2*r]   = (__bf16)bp1[kk*ldb];     bf1[2*r+1] = (__bf16)bp1[(kk+1)*ldb];
            }
        }
        acc00 = __builtin_amdgcn_wmma_f32_16x16x32_bf16(false, af0, false, bf0,
                                                        (short)0, acc00, false, false);
        acc01 = __builtin_amdgcn_wmma_f32_16x16x32_bf16(false, af0, false, bf1,
                                                        (short)0, acc01, false, false);
        acc10 = __builtin_amdgcn_wmma_f32_16x16x32_bf16(false, af1, false, bf0,
                                                        (short)0, acc10, false, false);
        acc11 = __builtin_amdgcn_wmma_f32_16x16x32_bf16(false, af1, false, bf1,
                                                        (short)0, acc11, false, false);
    }

    #pragma unroll
    for (int r = 0; r < 8; ++r) {
        int mm0 = m0 + r + 8*hlf;          // C/D layout: vgpr r -> M=r / M=r+8 per half-wave
        int mm1 = mm0 + 16;
        int c00 = mm0*ldc + n0 + mr, c01 = c00 + 16;
        int c10 = mm1*ldc + n0 + mr, c11 = c10 + 16;
        float v00 = alpha * acc00[r], v01 = alpha * acc01[r];
        float v10 = alpha * acc10[r], v11 = alpha * acc11[r];
        if (Ci) {
            v00 += beta * Ci[c00]; v01 += beta * Ci[c01];
            v10 += beta * Ci[c10]; v11 += beta * Ci[c11];
        }
        Cb[c00] = v00; Cb[c01] = v01; Cb[c10] = v10; Cb[c11] = v11;
    }
}

// ---------------- helpers ----------------
__device__ inline float blockReduceSum256(float v, float* sdata) {
    int tid = threadIdx.x;
    sdata[tid] = v; __syncthreads();
    for (int s = 128; s > 0; s >>= 1) { if (tid < s) sdata[tid] += sdata[tid+s]; __syncthreads(); }
    float r = sdata[0]; __syncthreads(); return r;
}

// rmsnorm(seq) * g, for both store and retrieve gammas (shared rsqrt)
__global__ __launch_bounds__(256) void k_rmsnorm_dual(
    const float* __restrict__ seq, const float* __restrict__ g1, const float* __restrict__ g2,
    float* __restrict__ o1, float* __restrict__ o2)
{
    __shared__ float sdata[256];
    size_t base = (size_t)blockIdx.x * DIMM;
    int t = threadIdx.x;
    float x0 = seq[base + t], x1 = seq[base + 256 + t];
    float ssum = blockReduceSum256(x0*x0 + x1*x1, sdata);
    float r = rsqrtf(ssum * (1.0f/DIMM) + 1e-6f);
    o1[base+t] = x0*r*g1[t];  o1[base+256+t] = x1*r*g1[256+t];
    o2[base+t] = x0*r*g2[t];  o2[base+256+t] = x1*r*g2[256+t];
}

// sigmoid(X @ W(512x4) + b); mode0: out[row*4+h]; mode1: out[(b*4+h)*32+ci]
__global__ void k_gemv4sig(const float* __restrict__ X, const float* __restrict__ W,
                           const float* __restrict__ bias, float* __restrict__ out,
                           int rows, int mode)
{
    int t = blockIdx.x * blockDim.x + threadIdx.x;
    if (t >= rows * HEADS) return;
    int row = t >> 2, h = t & 3;
    const float* x = X + (size_t)row * DIMM;
    float acc = bias ? bias[h] : 0.0f;
    for (int j = 0; j < DIMM; ++j) acc += x[j] * W[j*HEADS + h];
    float sgm = 1.0f / (1.0f + __expf(-acc));
    if (mode == 0) out[(size_t)row*HEADS + h] = sgm;
    else { int b = row / NCHNK, ci = row % NCHNK; out[((size_t)(b*HEADS+h))*NCHNK + ci] = sgm; }
}

__global__ void k_pooled(const float* __restrict__ S, float* __restrict__ P) {
    int bc = blockIdx.x;                 // b*NC + ci
    int b = bc / NCHNK, ci = bc % NCHNK;
    int col = threadIdx.x;               // 512 threads
    size_t base = ((size_t)b*NTOK + (size_t)ci*CHK) * DIMM + col;
    float s = 0.f;
    for (int r = 0; r < CHK; ++r) s += S[base + (size_t)r*DIMM];
    P[(size_t)bc*DIMM + col] = s * (1.0f/CHK);
}

__global__ void k_transpose_w2(const float* __restrict__ W2, float* __restrict__ W2T) {
    int t = blockIdx.x*blockDim.x + threadIdx.x;
    if (t >= HEADS*HIDD*DHH) return;
    int h = t / (HIDD*DHH); int rem = t % (HIDD*DHH);
    int j = rem / DHH; int d = rem % DHH;
    W2T[((size_t)h*DHH + d)*HIDD + j] = W2[t];
}

__global__ void k_gelu(const float* __restrict__ in, float* __restrict__ out, size_t n) {
    size_t i = (size_t)blockIdx.x*blockDim.x + threadIdx.x;
    if (i >= n) return;
    float x = in[i];
    out[i] = 0.5f * x * (1.0f + erff(x * 0.70710678118654752f));
}

__global__ void k_gelu_bwd(float* __restrict__ g, const float* __restrict__ pre, size_t n) {
    size_t i = (size_t)blockIdx.x*blockDim.x + threadIdx.x;
    if (i >= n) return;
    float x = pre[i];
    float cdf = 0.5f * (1.0f + erff(x * 0.70710678118654752f));
    float pdf = 0.3989422804014327f * __expf(-0.5f*x*x);
    g[i] *= (cdf + x*pdf);
}

// per-chunk: rmsnorm fwd, weighted-MSE grad, rmsnorm bwd -> dh2 (64x128), sg = -dgamma (128)
__global__ __launch_bounds__(256) void k_lossgrad(
    const float* __restrict__ H2, const float* __restrict__ K, const float* __restrict__ V,
    const float* __restrict__ LR, const float* __restrict__ mgamma,
    float* __restrict__ DH2, float* __restrict__ GG)
{
    __shared__ float red[64];
    __shared__ float gl[DHH];
    int z = blockIdx.x;
    int ci = z % NCHNK; int h = (z/NCHNK) % HEADS; int b = z / (NCHNK*HEADS);
    int tid = threadIdx.x;
    int row = tid >> 2, qq = tid & 3, col0 = qq*32;
    size_t h2base = (size_t)z*CHK*DHH + (size_t)row*DHH + col0;
    size_t tok    = (size_t)b*NTOK + (size_t)ci*CHK + row;
    size_t kvbase = tok*DIMM + (size_t)h*DHH + col0;
    float lrw = LR[tok*HEADS + h];
    const float* gm = mgamma + h*DHH + col0;

    if (tid < 64)  red[tid] = 0.f;
    if (tid < DHH) gl[tid]  = 0.f;
    __syncthreads();

    float hv[32]; float ss = 0.f;
    #pragma unroll
    for (int i = 0; i < 32; ++i) { hv[i] = H2[h2base + i]; ss += hv[i]*hv[i]; }
    atomicAdd(&red[row], ss);
    __syncthreads();
    float r = rsqrtf(red[row]*(1.0f/DHH) + 1e-6f);
    __syncthreads();
    if (tid < 64) red[tid] = 0.f;
    __syncthreads();

    float dyv[32]; float dot = 0.f;
    float coef = 2.0f * lrw * (1.0f/DHH);
    #pragma unroll
    for (int i = 0; i < 32; ++i) {
        float y = hv[i]*r;
        float g = gm[i];
        float e = y*g + K[kvbase+i] - V[kvbase+i];
        float dpred = e * coef;
        atomicAdd(&gl[col0+i], dpred*y);
        float dy = dpred * g;
        dyv[i] = dy;
        dot += dy*hv[i];
    }
    atomicAdd(&red[row], dot);
    __syncthreads();
    float dsum = red[row];
    float r3 = r*r*r*(1.0f/DHH);
    #pragma unroll
    for (int i = 0; i < 32; ++i)
        DH2[h2base + i] = r*dyv[i] - r3*hv[i]*dsum;
    __syncthreads();
    if (tid < DHH) GG[(size_t)z*DHH + tid] = -gl[tid];   // surprise = -grad
}

// Newton-Schulz prologue: t = -g / max(||g||_F, 1e-7), in place (one matrix per block)
__global__ __launch_bounds__(256) void k_nsnorm(float* __restrict__ T) {
    __shared__ float sdata[256];
    size_t base = (size_t)blockIdx.x * NMAT;
    int tid = threadIdx.x;
    float ss = 0.f;
    for (int i = tid; i < NMAT; i += 256) { float v = T[base+i]; ss += v*v; }
    float tot = blockReduceSum256(ss, sdata);
    float scale = -1.0f / fmaxf(sqrtf(tot), 1e-7f);
    for (int i = tid; i < NMAT; i += 256) T[base+i] *= scale;
}

// momentum + decay scans over chunks; writes SHIFTED weights (r_t = u_{t-1}, r_0 = Winit)
__global__ void k_scanmat(const float* __restrict__ S, float* __restrict__ R,
                          const float* __restrict__ MOMp, const float* __restrict__ DECp,
                          const float* __restrict__ Winit)
{
    int bh = blockIdx.y; int h = bh & 3;
    int e = blockIdx.x*blockDim.x + threadIdx.x;
    float u = Winit[(size_t)h*NMAT + e];
    float m = 0.f;
    for (int ci = 0; ci < NCHNK; ++ci) {
        size_t idx = ((size_t)(bh*NCHNK + ci))*NMAT + e;
        float s = S[idx];
        m = MOMp[bh*NCHNK + ci]*m + s;
        R[idx] = u;
        u = (1.0f - DECp[bh*NCHNK + ci])*u + m;
    }
}

__global__ void k_scangamma(float* __restrict__ G, const float* __restrict__ MOMp,
                            const float* __restrict__ DECp, const float* __restrict__ mgamma)
{
    int bh = blockIdx.x; int h = bh & 3;
    int e = threadIdx.x;
    float u = mgamma[h*DHH + e];
    float m = 0.f;
    for (int ci = 0; ci < NCHNK; ++ci) {
        size_t idx = ((size_t)(bh*NCHNK+ci))*DHH + e;
        float s = G[idx];            // already -gg
        m = MOMp[bh*NCHNK+ci]*m + s;
        G[idx] = u;                  // shifted gamma for retrieval
        u = (1.0f - DECp[bh*NCHNK+ci])*u + m;
    }
}

// retrieval epilogue: rmsnorm(h2r, r_gamma) + q, * gate, assemble (b,n,h*DH+d)
__global__ __launch_bounds__(256) void k_retout(
    const float* __restrict__ H2, const float* __restrict__ RG,
    const float* __restrict__ Q, const float* __restrict__ GATE, float* __restrict__ OUTB)
{
    __shared__ float red[64];
    int z = blockIdx.x;
    int ci = z % NCHNK; int h = (z/NCHNK)%HEADS; int b = z/(NCHNK*HEADS);
    int tid = threadIdx.x;
    int row = tid>>2, qq = tid&3, col0 = qq*32;
    size_t h2base = (size_t)z*CHK*DHH + (size_t)row*DHH + col0;
    size_t tok = (size_t)b*NTOK + (size_t)ci*CHK + row;
    size_t qbase = tok*DIMM + (size_t)h*DHH + col0;
    if (tid<64) red[tid]=0.f;
    __syncthreads();
    float hv[32]; float ss=0.f;
    #pragma unroll
    for (int i=0;i<32;++i){ hv[i]=H2[h2base+i]; ss+=hv[i]*hv[i]; }
    atomicAdd(&red[row], ss);
    __syncthreads();
    float r = rsqrtf(red[row]*(1.0f/DHH) + 1e-6f);
    float gt = GATE[tok*HEADS + h];
    #pragma unroll
    for (int i=0;i<32;++i){
        float o = hv[i]*r*RG[(size_t)z*DHH + col0 + i] + Q[qbase+i];
        OUTB[qbase+i] = o*gt;
    }
}

// ---------------- host orchestration ----------------
extern "C" void kernel_launch(void* const* d_in, const int* in_sizes, int n_in,
                              void* d_out, int out_size, void* d_ws, size_t ws_size,
                              hipStream_t stream) {
    (void)in_sizes; (void)n_in; (void)out_size;
    if (ws_size < WS_FLOATS * sizeof(float)) return;

    const float* seq     = (const float*)d_in[0];
    const float* store_g = (const float*)d_in[1];
    const float* retr_g  = (const float*)d_in[2];
    const float* Wq      = (const float*)d_in[3];
    const float* Wk      = (const float*)d_in[4];
    const float* Wv      = (const float*)d_in[5];
    const float* W_lr    = (const float*)d_in[6];
    const float* b_lr    = (const float*)d_in[7];
    const float* Wm      = (const float*)d_in[8];
    const float* bm      = (const float*)d_in[9];
    const float* Wd      = (const float*)d_in[10];
    const float* bd      = (const float*)d_in[11];
    const float* Wgate   = (const float*)d_in[12];
    const float* Wc      = (const float*)d_in[13];
    const float* mw1     = (const float*)d_in[14];
    const float* mw2     = (const float*)d_in[15];
    const float* mgamma  = (const float*)d_in[16];
    float* WS  = (float*)d_ws;
    float* out = (float*)d_out;

    auto gemm = [&](int M, int N, int Kd,
                    const float* A, int lda, int sAb, int sAh, int sAc, int tA,
                    const float* B, int ldb, int sBb, int sBh, int sBc, int tB,
                    float* C, int ldc, int sCb, int sCh, int sCc,
                    const float* Cin, float alpha, float beta, int nb, int nh, int nc) {
        dim3 grid(1, (unsigned)((M/32)*(N/32)/8), (unsigned)(nb*nh*nc));
        if (!tA && !tB)
            gemm_bf16<0,0><<<grid,256,0,stream>>>(M,N,Kd, A,lda,sAb,sAh,sAc,
                B,ldb,sBb,sBh,sBc, C,ldc,sCb,sCh,sCc, Cin, alpha,beta, nh,nc);
        else if (!tA && tB)
            gemm_bf16<0,1><<<grid,256,0,stream>>>(M,N,Kd, A,lda,sAb,sAh,sAc,
                B,ldb,sBb,sBh,sBc, C,ldc,sCb,sCh,sCc, Cin, alpha,beta, nh,nc);
        else if (tA && !tB)
            gemm_bf16<1,0><<<grid,256,0,stream>>>(M,N,Kd, A,lda,sAb,sAh,sAc,
                B,ldb,sBb,sBh,sBc, C,ldc,sCb,sCh,sCc, Cin, alpha,beta, nh,nc);
        else
            gemm_bf16<1,1><<<grid,256,0,stream>>>(M,N,Kd, A,lda,sAb,sAh,sAc,
                B,ldb,sBb,sBh,sBc, C,ldc,sCb,sCh,sCc, Cin, alpha,beta, nh,nc);
    };

    // 1) dual rmsnorm
    k_rmsnorm_dual<<<BBATCH*NTOK, 256, 0, stream>>>(seq, store_g, retr_g,
                                                    WS+OFF_SSTORE, WS+OFF_SRETR);
    // 2) big projections (M=4096, N=512, K=512)
    gemm(4096,512,512, WS+OFF_SSTORE,512,0,0,0,0, Wk,512,0,0,0,0,
         WS+OFF_K,512,0,0,0, nullptr,1.f,0.f, 1,1,1);
    gemm(4096,512,512, WS+OFF_SSTORE,512,0,0,0,0, Wv,512,0,0,0,0,
         WS+OFF_V,512,0,0,0, nullptr,1.f,0.f, 1,1,1);
    gemm(4096,512,512, WS+OFF_SRETR,512,0,0,0,0, Wq,512,0,0,0,0,
         WS+OFF_Q,512,0,0,0, nullptr,1.f,0.f, 1,1,1);
    // 3) small sigmoid projections, pooling, mw2 transpose
    k_gemv4sig<<<(4096*HEADS+255)/256,256,0,stream>>>(WS+OFF_SSTORE, W_lr, b_lr, WS+OFF_LR, 4096, 0);
    k_gemv4sig<<<(4096*HEADS+255)/256,256,0,stream>>>(WS+OFF_SRETR, Wgate, nullptr, WS+OFF_GATE, 4096, 0);
    k_pooled<<<BBATCH*NCHNK, 512, 0, stream>>>(WS+OFF_SSTORE, WS+OFF_POOLED);
    k_gemv4sig<<<1,256,0,stream>>>(WS+OFF_POOLED, Wm, bm, WS+OFF_MOM, BBATCH*NCHNK, 1);
    k_gemv4sig<<<1,256,0,stream>>>(WS+OFF_POOLED, Wd, bd, WS+OFF_DEC, BBATCH*NCHNK, 1);
    k_transpose_w2<<<(HEADS*HIDD*DHH+255)/256,256,0,stream>>>(mw2, WS+OFF_MW2T);
    // 4) memory-MLP forward on k: h1 = k @ mw1 ; a1 = gelu(h1) ; h2 = a1 @ mw2
    gemm(64,512,128, WS+OFF_K,512, 1048576,128,32768,0,
         mw1,512, 0,65536,0,0,
         WS+OFF_H1,512, 4194304,1048576,32768, nullptr,1.f,0.f, 2,4,32);
    k_gelu<<<8388608/256,256,0,stream>>>(WS+OFF_H1, WS+OFF_A1, (size_t)8388608);
    gemm(64,128,512, WS+OFF_A1,512, 4194304,1048576,32768,0,
         mw2,128, 0,65536,0,0,
         WS+OFF_H2,128, 1048576,262144,8192, nullptr,1.f,0.f, 2,4,32);
    // 5) loss backward through rmsnorm -> dh2, sg
    k_lossgrad<<<256,256,0,stream>>>(WS+OFF_H2, WS+OFF_K, WS+OFF_V, WS+OFF_LR, mgamma,
                                     WS+OFF_DH2, WS+OFF_GG);
    // 6) dW2^T = dh2^T @ a1 -> S12 half1 (kept transposed, 128x512)
    gemm(128,512,64, WS+OFF_DH2,128, 1048576,262144,8192,1,
         WS+OFF_A1,512, 4194304,1048576,32768,0,
         WS+OFF_S12+16777216,512, 8388608,2097152,65536, nullptr,1.f,0.f, 2,4,32);
    // 7) da1 = dh2 @ mw2^T -> A1 ; dh1 = da1 * gelu'(h1)
    gemm(64,512,128, WS+OFF_DH2,128, 1048576,262144,8192,0,
         mw2,128, 0,65536,0,1,
         WS+OFF_A1,512, 4194304,1048576,32768, nullptr,1.f,0.f, 2,4,32);
    k_gelu_bwd<<<8388608/256,256,0,stream>>>(WS+OFF_A1, WS+OFF_H1, (size_t)8388608);
    // 8) dW1 = kc^T @ dh1 -> S12 half0
    gemm(128,512,64, WS+OFF_K,512, 1048576,128,32768,1,
         WS+OFF_A1,512, 4194304,1048576,32768,0,
         WS+OFF_S12,512, 8388608,2097152,65536, nullptr,1.f,0.f, 2,4,32);

    // 9) Newton-Schulz (5 iters) on 256 matrices, ping-pong with T2 (= H1..A1 region)
    auto ns_half = [&](float* t0) {
        float* t1 = WS + OFF_H1;  // T2 (16777216 floats)
        k_nsnorm<<<256,256,0,stream>>>(t0);
        for (int step = 0; step < 5; ++step) {
            float* tin  = (step & 1) ? t1 : t0;
            float* tout = (step & 1) ? t0 : t1;
            // A = t @ t^T
            gemm(128,128,512, tin,512, 0,0,65536,0, tin,512, 0,0,65536,1,
                 WS+OFF_ABUF,128, 0,0,16384, nullptr, 1.f,0.f, 1,1,256);
            // Bm = c*(A@A) + b*A
            gemm(128,128,128, WS+OFF_ABUF,128, 0,0,16384,0, WS+OFF_ABUF,128, 0,0,16384,0,
                 WS+OFF_BM,128, 0,0,16384, WS+OFF_ABUF, 2.0315f,-4.775f, 1,1,256);
            // t' = Bm@t + a*t
            gemm(128,512,128, WS+OFF_BM,128, 0,0,16384,0, tin,512, 0,0,65536,0,
                 tout,512, 0,0,65536, tin, 1.f,3.4445f, 1,1,256);
        }
        // result in t1 (T2)
    };
    ns_half(WS+OFF_S12);
    k_scanmat<<<dim3(256,8),256,0,stream>>>(WS+OFF_H1, WS+OFF_S12,
                                            WS+OFF_MOM, WS+OFF_DEC, mw1);
    ns_half(WS+OFF_S12+16777216);
    k_scanmat<<<dim3(256,8),256,0,stream>>>(WS+OFF_H1, WS+OFF_S12+16777216,
                                            WS+OFF_MOM, WS+OFF_DEC, WS+OFF_MW2T);
    k_scangamma<<<8,128,0,stream>>>(WS+OFF_GG, WS+OFF_MOM, WS+OFF_DEC, mgamma);

    // 10) retrieval: h1r = q @ r1 ; a1r = gelu ; h2r = a1r @ r2t^T
    gemm(64,512,128, WS+OFF_Q,512, 1048576,128,32768,0,
         WS+OFF_S12,512, 8388608,2097152,65536,0,
         WS+OFF_H1,512, 4194304,1048576,32768, nullptr,1.f,0.f, 2,4,32);
    k_gelu<<<8388608/256,256,0,stream>>>(WS+OFF_H1, WS+OFF_H1, (size_t)8388608);
    gemm(64,128,512, WS+OFF_H1,512, 4194304,1048576,32768,0,
         WS+OFF_S12+16777216,512, 8388608,2097152,65536,1,
         WS+OFF_H2,128, 1048576,262144,8192, nullptr,1.f,0.f, 2,4,32);
    // 11) rmsnorm + residual + gate, assemble heads
    k_retout<<<256,256,0,stream>>>(WS+OFF_H2, WS+OFF_GG, WS+OFF_Q, WS+OFF_GATE, WS+OFF_FINAL);
    // 12) output projection
    gemm(4096,512,512, WS+OFF_FINAL,512,0,0,0,0, Wc,512,0,0,0,0,
         out,512,0,0,0, nullptr,1.f,0.f, 1,1,1);
}
